// SGC_59528246723059
// MI455X (gfx1250) — matrix operand order repaired and checked by reading, not compiled, not run on hardware
//
#include <hip/hip_runtime.h>
#include <hip/hip_bf16.h>

// ---- CDNA5 WMMA vector types (gfx1250, wave32) ----
typedef __attribute__((ext_vector_type(16))) __bf16 v16bf;
typedef __attribute__((ext_vector_type(8)))  __bf16 v8bf;
typedef __attribute__((ext_vector_type(8)))  float  v8f;

#define DIN  256
#define KC   64             // K-chunk staged in LDS
#define LDSK (KC + 8)       // row stride 72 bf16 = 36 dwords -> bank-conflict-free
#define NB   128            // block/wave N footprint (8 x 16-col tiles)

// deg[i] = 1.0 (self-loop weight)
__global__ void k_deg_init(float* __restrict__ deg, int n) {
    int i = blockIdx.x * blockDim.x + threadIdx.x;
    if (i < n) deg[i] = 1.0f;
}

// deg[col[e]] += w[e]
__global__ void k_deg_scatter(const int* __restrict__ col, const float* __restrict__ w,
                              float* __restrict__ deg, int e) {
    int i = blockIdx.x * blockDim.x + threadIdx.x;
    if (i < e) atomicAdd(&deg[col[i]], w[i]);
}

// deg -> dinv in place
__global__ void k_dinv(float* __restrict__ deg, int n) {
    int i = blockIdx.x * blockDim.x + threadIdx.x;
    if (i < n) {
        float d = deg[i];
        deg[i] = (d > 0.0f) ? rsqrtf(d) : 0.0f;
    }
}

// agg[i,:] = dinv[i]^2 * x[i,:]   (self-loop contribution, also zero-initializes agg)
__global__ void k_selfloop_init(const float* __restrict__ x, const float* __restrict__ dinv,
                                float* __restrict__ agg, long long total) {
    long long t = (long long)blockIdx.x * blockDim.x + threadIdx.x;
    if (t < total) {
        int i = (int)(t >> 8);              // / DIN (DIN==256)
        float di = dinv[i];
        agg[t] = di * di * x[t];
    }
}

// One wave32 per edge: agg[col,:] += dinv[row]*w*dinv[col] * x[row,:]
// Lane-striped -> 8 fully-coalesced f32 atomics per lane; x and agg (51 MB each)
// are resident in the 192 MB L2, so the gather+atomics stay on-chip.
__global__ void k_edge_agg(const float* __restrict__ x,
                           const int* __restrict__ rowi, const int* __restrict__ coli,
                           const float* __restrict__ w, const float* __restrict__ dinv,
                           float* __restrict__ agg, int e) {
    long long gid = (long long)blockIdx.x * blockDim.x + threadIdx.x;
    int eid  = (int)(gid >> 5);
    int lane = threadIdx.x & 31;
    if (eid >= e) return;
    int j = rowi[eid];                      // source
    int i = coli[eid];                      // target
    float norm = dinv[j] * w[eid] * dinv[i];
    const float* xs = x   + (size_t)j * DIN;
    float*       ad = agg + (size_t)i * DIN;
#pragma unroll
    for (int t = 0; t < DIN / 32; ++t) {
        int d = lane + t * 32;
        atomicAdd(ad + d, norm * xs[d]);
    }
}

__device__ inline void cvt8(v16bf& v, int base, float4 p, float4 q) {
    v[base + 0] = (__bf16)p.x; v[base + 1] = (__bf16)p.y;
    v[base + 2] = (__bf16)p.z; v[base + 3] = (__bf16)p.w;
    v[base + 4] = (__bf16)q.x; v[base + 5] = (__bf16)q.y;
    v[base + 6] = (__bf16)q.z; v[base + 7] = (__bf16)q.w;
}

// 16B-aligned LDS fragment load (two ds_load_b128)
__device__ inline v16bf load_b_frag(const __bf16* p) {
    v8bf lo = *(const v8bf*)p;
    v8bf hi = *(const v8bf*)(p + 16);
    v16bf r;
#pragma unroll
    for (int i = 0; i < 8; ++i) { r[i] = lo[i]; r[i + 8] = hi[i]; }
    return r;
}

// C[M,Nn] = act(A[M,K] @ W[Nn,K]^T + bias[Nn])
// Block = 256 threads (8 waves), block tile = 128(M) x 128(N):
//   wave w owns M-tile w and all 8 N-tiles -> 8 f32 accumulators, 8 WMMAs per A frag.
//   W slab (128 rows x KC cols) is staged f32->bf16 into LDS once per block per chunk.
// Fragment layouts per CDNA5 ISA 7.12.2 (wave32):
//   A 16x32 bf16 : lane&15 = row; hi-half lanes K-offset +8; VGPRs 0-3 hold K=abase+0..7,
//                  VGPRs 4-7 hold K=abase+16..23 (contiguous pairs).
//   B 32x16 bf16 : lane&15 = col; hi-half lanes K-offset +8; K=bbase+0..7, bbase+16..23 (same as A).
//   C 16x16 f32  : lane&15 = col; VGPR r = row (hi half +8).
template <int RELU>
__global__ __launch_bounds__(256)
void k_gemm_bias_act(const float* __restrict__ A, const float* __restrict__ W,
                     const float* __restrict__ bias, float* __restrict__ C,
                     int M, int Nn, int K) {
    __shared__ __bf16 ldsB[NB * LDSK];      // 128 * 72 * 2B = 18 KB

    const int lane = threadIdx.x & 31;
    const int wv   = threadIdx.x >> 5;      // 0..7
    const int hi   = lane >> 4;
    const int lr   = lane & 15;

    const int Mtiles = M >> 4;
    int mtile = blockIdx.x * 8 + wv;
    const bool valid = (mtile < Mtiles);
    if (!valid) mtile = Mtiles - 1;         // clamp loads; stay barrier-uniform
    const int nblk = blockIdx.y;            // 128-column block

    const float* arow = A + (size_t)(mtile * 16 + lr) * K;
    const int abase = hi * 8;
    const int bbase = hi * 8;

    v8f acc[8];
#pragma unroll
    for (int t = 0; t < 8; ++t) acc[t] = (v8f){};

    for (int k0 = 0; k0 < K; k0 += KC) {
        // ---- cooperative stage: W rows [nblk*NB, +NB), cols [k0, k0+KC) -> LDS bf16 ----
        for (int u = threadIdx.x; u < NB * (KC / 4); u += 256) {
            int r  = u / (KC / 4);
            int c4 = (u % (KC / 4)) * 4;
            float4 f = *(const float4*)(W + (size_t)(nblk * NB + r) * K + k0 + c4);
            __bf16* dst = ldsB + r * LDSK + c4;
            dst[0] = (__bf16)f.x; dst[1] = (__bf16)f.y;
            dst[2] = (__bf16)f.z; dst[3] = (__bf16)f.w;
        }
        __syncthreads();

        if (k0 + KC < K) __builtin_prefetch(arow + k0 + KC, 0, 1);

#pragma unroll
        for (int kk = 0; kk < KC; kk += 32) {
            // A fragment from global (f32 -> bf16 in-register), reused by 8 WMMAs
            float4 a0 = *(const float4*)(arow + k0 + kk + abase + 0);
            float4 a1 = *(const float4*)(arow + k0 + kk + abase + 4);
            float4 a2 = *(const float4*)(arow + k0 + kk + abase + 16);
            float4 a3 = *(const float4*)(arow + k0 + kk + abase + 20);
            v16bf af;
            cvt8(af, 0, a0, a1);
            cvt8(af, 8, a2, a3);

#pragma unroll
            for (int t = 0; t < 8; ++t) {
                v16bf bf = load_b_frag(ldsB + (t * 16 + lr) * LDSK + kk + bbase);
                acc[t] = __builtin_amdgcn_wmma_f32_16x16x32_bf16(
                    false, af, false, bf, (short)0, acc[t], false, false);
            }
        }
        __syncthreads();
    }

    if (!valid) return;

    const int row0 = mtile * 16 + hi * 8;
    const int col0 = nblk * NB + lr;
#pragma unroll
    for (int t = 0; t < 8; ++t) {
        const int col = col0 + t * 16;
        const float bs = bias[col];
#pragma unroll
        for (int r = 0; r < 8; ++r) {
            float v = acc[t][r] + bs;
            if (RELU) v = fmaxf(v, 0.0f);
            C[(size_t)(row0 + r) * Nn + col] = v;
        }
    }
}

extern "C" void kernel_launch(void* const* d_in, const int* in_sizes, int n_in,
                              void* d_out, int out_size, void* d_ws, size_t ws_size,
                              hipStream_t stream) {
    const float* x  = (const float*)d_in[0];
    const int*   ei = (const int*)d_in[1];   // edge_index [2,E] (int, per harness convention)
    const float* ew = (const float*)d_in[2];
    const float* W1 = (const float*)d_in[3];
    const float* b1 = (const float*)d_in[4];
    const float* W2 = (const float*)d_in[5];
    const float* b2 = (const float*)d_in[6];

    const int n    = in_sizes[0] / DIN;   // 50000
    const int e    = in_sizes[2];         // 1,000,000
    const int dhid = in_sizes[4];         // 512
    const int dout = in_sizes[6];         // 256

    // workspace: [deg/dinv : n f32 (padded)] [agg : n*DIN f32]
    float* deg = (float*)d_ws;
    float* agg = deg + (((size_t)n + 255) & ~(size_t)255);

    float* logits = (float*)d_out;                      // [n, dout]
    float* hbuf   = logits + (size_t)n * dout;          // [n, dhid] (2nd output, reused as GEMM2 input)

    const int* rowi = ei;                               // edge_index[0] = source j
    const int* coli = ei + e;                           // edge_index[1] = target i

    // 1) degree with self-loops -> dinv
    k_deg_init<<<(n + 255) / 256, 256, 0, stream>>>(deg, n);
    k_deg_scatter<<<(e + 255) / 256, 256, 0, stream>>>(coli, ew, deg, e);
    k_dinv<<<(n + 255) / 256, 256, 0, stream>>>(deg, n);

    // 2) aggregation: self-loop init + one wave per edge
    long long tot = (long long)n * DIN;
    k_selfloop_init<<<(int)((tot + 255) / 256), 256, 0, stream>>>(x, deg, agg, tot);
    long long thr = (long long)e * 32;
    k_edge_agg<<<(int)((thr + 255) / 256), 256, 0, stream>>>(x, rowi, coli, ew, deg, agg, e);

    // 3) h = relu(agg @ W1^T + b1)   [WMMA bf16 -> f32, LDS-staged weights]
    {
        dim3 grid((n / 16 + 7) / 8, dhid / NB);
        k_gemm_bias_act<1><<<grid, 256, 0, stream>>>(agg, W1, b1, hbuf, n, dhid, DIN);
    }
    // 4) logits = h @ W2^T + b2
    {
        dim3 grid((n / 16 + 7) / 8, dout / NB);
        k_gemm_bias_act<0><<<grid, 256, 0, stream>>>(hbuf, W2, b2, logits, n, dout, dhid);
    }
}
